// MoME_89515708383723
// MI455X (gfx1250) — matrix operand order repaired and checked by reading, not compiled, not run on hardware
//
#include <hip/hip_runtime.h>
#include <math.h>
#include <stdint.h>

// Problem constants (from reference)
#define BGRAPHS 1024
#define NNODES  262144
#define DFEAT   128
#define NEXP    8
#define ODIM    128

typedef _Float16 half_t;
typedef __attribute__((ext_vector_type(16))) _Float16 v16h;
typedef __attribute__((ext_vector_type(8)))  _Float16 v8h;
typedef __attribute__((ext_vector_type(8)))  float    v8f;
typedef __attribute__((ext_vector_type(4)))  unsigned int v4u;
typedef __attribute__((ext_vector_type(8)))  int          v8i;
typedef __attribute__((ext_vector_type(4)))  int          v4i;

// Workspace layout (bytes)
#define FG_OFF     0u        // feat_graphs  [1024][128] f32  (512 KB)
#define GW_OFF     524288u   // gates/E      [1024][8]   f32  (32 KB)
#define IMP_OFF    557056u   // importance   [8] f32
#define LOADV_OFF  557088u   // load         [8] f32
#define WT_OFF     557120u   // w_experts^T f16 [8][128][136]  (272 KB)
#define WT_STRIDE  136       // halfs per o-row (128 + 8 pad)

// Dynamic LDS layout for k_expert (bytes)
#define AH_OFF    0u         // [64][136] f16 features          (17408)
#define WROW_OFF  17408u     // [64][8] f32 per-row gate/E      (2048)
#define SEG_OFF   19456u     // [64] i32                        (256)
#define ACT_OFF   19712u     // [8] i32 active flags            (32)
#define IDX_OFF   19744u     // [8] i32 active expert list      (32)
#define NACT_OFF  19776u     // i32 count                       (pad to 16B)
#define BT0_OFF   19792u     // [128][136] f16 weights buf 0    (34816)
#define BT1_OFF   54608u     // [128][136] f16 weights buf 1    (34816)
#define SMEM_BYTES 89424u
#define WT_TILE_BYTES 34816u // one expert weight tile
#define WT_TILE_E8    4352u  // ... in 8-byte elements

// ---------------------------------------------------------------------------
// Zero accumulators: feat_graphs, y (d_out), importance+load
// ---------------------------------------------------------------------------
__global__ void k_zero(float* __restrict__ fg, float* __restrict__ y,
                       float* __restrict__ stats) {
    int i = blockIdx.x * 256 + threadIdx.x;
    if (i < BGRAPHS * DFEAT) { fg[i] = 0.f; y[i] = 0.f; }
    if (i < 16) stats[i] = 0.f;
}

// ---------------------------------------------------------------------------
// Convert w_experts f32 [E][D][O] -> f16 transposed [E][O][K=D], padded stride
// ---------------------------------------------------------------------------
__global__ void k_prepw(const float* __restrict__ we, half_t* __restrict__ wt) {
    int i = blockIdx.x * 256 + threadIdx.x;
    if (i < NEXP * DFEAT * ODIM) {
        int e = i >> 14, rem = i & 16383, d = rem >> 7, o = rem & 127;
        wt[(size_t)e * DFEAT * WT_STRIDE + o * WT_STRIDE + d] = (half_t)we[i];
    }
}

// ---------------------------------------------------------------------------
// Ragged per-graph sum pooling (segment_ids sorted): 128 threads = one per dim,
// 128 nodes per block; flush on segment change via f32 global atomics.
// ---------------------------------------------------------------------------
__global__ void k_pool(const float* __restrict__ feat, const int* __restrict__ seg,
                       float* __restrict__ fg) {
    __shared__ int s[128];
    int d = threadIdx.x;
    int base = blockIdx.x * 128;
    s[d] = seg[base + d];
    __syncthreads();
    float acc = 0.f;
    int cur = s[0];
    for (int i = 0; i < 128; i++) {
        int si = s[i];                      // uniform across threads
        if (si != cur) { atomicAdd(&fg[(size_t)cur * DFEAT + d], acc); acc = 0.f; cur = si; }
        acc += feat[(size_t)(base + i) * DFEAT + d];
    }
    atomicAdd(&fg[(size_t)cur * DFEAT + d], acc);
}

// ---------------------------------------------------------------------------
// Noisy top-k gating, one thread per graph row (full f32 — top-k is discrete).
// ---------------------------------------------------------------------------
__global__ void k_gate(const float* __restrict__ fg, const float* __restrict__ noise,
                       const float* __restrict__ wgate, const float* __restrict__ wnoise,
                       float* __restrict__ gout, float* __restrict__ gw,
                       float* __restrict__ imp, float* __restrict__ loadv) {
    __shared__ float wgS[DFEAT * NEXP];
    __shared__ float wnS[DFEAT * NEXP];
    int tid = threadIdx.x;
    for (int i = tid; i < DFEAT * NEXP; i += 256) { wgS[i] = wgate[i]; wnS[i] = wnoise[i]; }
    __syncthreads();

    int b = blockIdx.x * 256 + tid;   // grid is exactly 1024 threads
    float cl[NEXP], nz[NEXP];
#pragma unroll
    for (int e = 0; e < NEXP; e++) { cl[e] = 0.f; nz[e] = 0.f; }
    const float* fr = fg + (size_t)b * DFEAT;
    for (int d = 0; d < DFEAT; d++) {
        float f = fr[d];
#pragma unroll
        for (int e = 0; e < NEXP; e++) {
            cl[e] += f * wgS[d * NEXP + e];
            nz[e] += f * wnS[d * NEXP + e];
        }
    }
    float stdv[NEXP], ny[NEXP];
#pragma unroll
    for (int e = 0; e < NEXP; e++) {
        float x = nz[e];
        float sp = (x > 20.f) ? x : log1pf(expf(x));   // softplus
        stdv[e] = sp + 0.01f;
        ny[e] = cl[e] + noise[(size_t)b * NEXP + e] * stdv[e];
    }
    // top-3 of noisy logits (descending, first index wins ties like lax.top_k)
    float v0 = -3.402823e38f, v1 = v0, v2 = v0;
    int i0 = 0, i1 = 0;
#pragma unroll
    for (int e = 0; e < NEXP; e++) {
        float v = ny[e];
        if (v > v0)      { v2 = v1; v1 = v0; i1 = i0; v0 = v; i0 = e; }
        else if (v > v1) { v2 = v1; v1 = v; i1 = e; }
        else if (v > v2) { v2 = v; }
    }
    // softmax over top-2
    float e1 = expf(v1 - v0);
    float inv = 1.f / (1.f + e1);
    float g0 = inv, g1 = e1 * inv;

    float gr[NEXP];
#pragma unroll
    for (int e = 0; e < NEXP; e++) gr[e] = 0.f;
    gr[i0] = g0; gr[i1] = g1;
#pragma unroll
    for (int e = 0; e < NEXP; e++) {
        gout[(size_t)b * NEXP + e] = gr[e];
        gw[(size_t)b * NEXP + e] = gr[e] * 0.125f;   // fold mean over E=8
    }
    atomicAdd(&imp[i0], g0);
    atomicAdd(&imp[i1], g1);

    const float inv_sqrt2 = 0.70710678118654752f;
#pragma unroll
    for (int e = 0; e < NEXP; e++) {
        float thr = (ny[e] > v2) ? v2 : v1;     // thr_in = 3rd, thr_out = 2nd
        float z = (cl[e] - thr) / stdv[e];
        float p = 0.5f * (1.f + erff(z * inv_sqrt2));
        atomicAdd(&loadv[e], p);
    }
}

// ---------------------------------------------------------------------------
// gate_loss = cv2(importance) + cv2(load), unbiased variance (ddof=1)
// ---------------------------------------------------------------------------
__global__ void k_loss(const float* __restrict__ imp, const float* __restrict__ loadv,
                       float* __restrict__ out) {
    if (threadIdx.x == 0 && blockIdx.x == 0) {
        float mi = 0.f, ml = 0.f;
        for (int e = 0; e < NEXP; e++) { mi += imp[e]; ml += loadv[e]; }
        mi *= 0.125f; ml *= 0.125f;
        float vi = 0.f, vl = 0.f;
        for (int e = 0; e < NEXP; e++) {
            float di = imp[e] - mi;  vi += di * di;
            float dl = loadv[e] - ml; vl += dl * dl;
        }
        vi *= (1.f / 7.f); vl *= (1.f / 7.f);
        out[0] = vi / (mi * mi + 1e-10f) + vl / (ml * ml + 1e-10f);
    }
}

// ---------------------------------------------------------------------------
// TDM: async DMA one contiguous 34816-byte weight tile (4352 x 8B) into LDS.
// D# per CDNA5 ISA (group0: count/lds/global/type=2;
// group1: data_size=3(8B), tensor_dim0=tile_dim0=n8, dim1=1; groups2/3: zero).
// Tracked with TENSORcnt.  6-arg builtin form (clang-23 / therock headers).
// ---------------------------------------------------------------------------
__device__ __forceinline__ void tdm_load_1d(const void* gsrc, unsigned lds_byte,
                                            unsigned n8) {
    unsigned long long ga = (unsigned long long)(uintptr_t)gsrc;
    v4u g0;
    g0[0] = 1u;                                   // count=1, user descriptor
    g0[1] = lds_byte;                             // lds_addr
    g0[2] = (unsigned)ga;                         // global_addr[31:0]
    g0[3] = (unsigned)((ga >> 32) & 0x1FFFFFFull) // global_addr[56:32]
          | (2u << 30);                           // type=2 ("image")
    v8i g1;
    g1[0] = (int)(3u << 16);                      // wg_mask=0, data_size=3 (8B)
    g1[1] = (int)((n8 & 0xFFFFu) << 16);          // tensor_dim0[15:0]
    g1[2] = (int)(((n8 >> 16) & 0xFFFFu)          // tensor_dim0[31:16]
          | (1u << 16));                          // tensor_dim1 = 1
    g1[3] = (int)((n8 & 0xFFFFu) << 16);          // tile_dim0 = n8
    g1[4] = 1;                                    // tile_dim1 = 1, tile_dim2 = 0
    g1[5] = (int)n8;                              // tensor_dim0_stride[31:0]
    g1[6] = 0;
    g1[7] = 0;
    v4i z4; z4[0] = 0; z4[1] = 0; z4[2] = 0; z4[3] = 0;
    v8i z8;
#pragma unroll
    for (int i = 0; i < 8; i++) z8[i] = 0;
    __builtin_amdgcn_tensor_load_to_lds(g0, g1, z4, z4, z8, 0);
}

// ---------------------------------------------------------------------------
// Fused expert GEMM + relu + gate-weighting + segment pooling.
// 256 threads (8 wave32), 64 nodes x 128 out cols, WMMA f16->f32.
// Active-expert weight tiles are TDM-prefetched into double-buffered LDS,
// overlapping DMA of expert j+1 with WMMA compute of expert j.
// ---------------------------------------------------------------------------
__global__ __launch_bounds__(256)
void k_expert(const float* __restrict__ feat, const int* __restrict__ seg,
              const float* __restrict__ gw, const half_t* __restrict__ wt,
              float* __restrict__ y) {
    extern __shared__ __align__(16) unsigned char smem[];
    half_t* Ah    = (half_t*)(smem + AH_OFF);
    float*  wrowS = (float*)(smem + WROW_OFF);
    int*    segsS = (int*)(smem + SEG_OFF);
    int*    actS  = (int*)(smem + ACT_OFF);
    int*    idxS  = (int*)(smem + IDX_OFF);
    int*    nactS = (int*)(smem + NACT_OFF);
    float*  outS  = (float*)(smem + BT0_OFF);     // alias of weight buf 0

    int tid  = threadIdx.x;
    int lane = tid & 31;
    int widx = tid >> 5;
    int rt   = widx >> 1;      // row-tile 0..3 (16 rows each)
    int cg   = widx & 1;       // column half 0..1 (64 cols each)
    int base = blockIdx.x * 64;

    // LDS byte offsets of the two weight buffers (aperture low 32 bits)
    unsigned bt_lds[2];
    bt_lds[0] = (unsigned)(uintptr_t)(smem + BT0_OFF);
    bt_lds[1] = (unsigned)(uintptr_t)(smem + BT1_OFF);

    // ---- stage 64x128 features f32 -> f16 into padded LDS rows ----
    const float4* fsrc = (const float4*)(feat + (size_t)base * DFEAT);
#pragma unroll
    for (int q = tid; q < 2048; q += 256) {
        float4 v = fsrc[q];
        int row = q >> 5, c4 = (q & 31) << 2;
        half_t* dst = Ah + row * WT_STRIDE + c4;
        dst[0] = (half_t)v.x; dst[1] = (half_t)v.y;
        dst[2] = (half_t)v.z; dst[3] = (half_t)v.w;
    }
    if (tid < 64) segsS[tid] = seg[base + tid];
    for (int i = tid; i < 512; i += 256) {
        int r = i >> 3;
        int s = seg[base + r];
        wrowS[i] = gw[(size_t)s * NEXP + (i & 7)];
    }
    if (blockIdx.x + 1 < gridDim.x)   // hint next feature chunk into cache
        __builtin_prefetch(feat + (size_t)(base + 64) * DFEAT + tid * 32, 0, 1);
    __syncthreads();

    // ---- which experts have any nonzero weight in this chunk? ----
    if (tid < NEXP) {
        int f = 0;
        for (int r = 0; r < 64; r++) f |= (wrowS[r * 8 + tid] != 0.f);
        actS[tid] = f;
    }
    __syncthreads();
    if (tid == 0) {
        int n = 0;
        for (int e = 0; e < NEXP; e++) if (actS[e]) idxS[n++] = e;
        nactS[0] = n;
    }

    // ---- A fragments (ISA 16-bit A layout: K=(h&7)+(h/8)*16+(lane/16)*8) ----
    v16h afr[4];
#pragma unroll
    for (int kc = 0; kc < 4; kc++) {
        int row = rt * 16 + (lane & 15);
        const half_t* p = Ah + row * WT_STRIDE + kc * 32 + ((lane >> 4) << 3);
        v8h lo = *(const v8h*)p;
        v8h hi = *(const v8h*)(p + 16);
#pragma unroll
        for (int h = 0; h < 8; h++) { afr[kc][h] = lo[h]; afr[kc][h + 8] = hi[h]; }
    }

    v8f acc[4];
#pragma unroll
    for (int i = 0; i < 4; i++)
#pragma unroll
        for (int j = 0; j < 8; j++) acc[i][j] = 0.f;

    __syncthreads();
    int nact = nactS[0];

    // ---- prologue: wave 0 kicks off DMA of the first expert's weights ----
    if (widx == 0 && nact > 0)
        tdm_load_1d(wt + (size_t)idxS[0] * DFEAT * WT_STRIDE, bt_lds[0], WT_TILE_E8);

    int rowbase = rt * 16 + ((lane >> 4) << 3);   // C-tile rows this lane owns

    for (int j = 0; j < nact; j++) {
        int e = idxS[j];
        if (widx == 0) {
            if (j + 1 < nact) {   // prefetch next expert into the other buffer
                tdm_load_1d(wt + (size_t)idxS[j + 1] * DFEAT * WT_STRIDE,
                            bt_lds[(j + 1) & 1], WT_TILE_E8);
                __builtin_amdgcn_s_wait_tensorcnt((short)1);  // tile j landed
            } else {
                __builtin_amdgcn_s_wait_tensorcnt((short)0);
            }
        }
        __syncthreads();          // publish tile j to all waves

        const half_t* BtS = (const half_t*)(smem + ((j & 1) ? BT1_OFF : BT0_OFF));

        float w8[8];
#pragma unroll
        for (int r = 0; r < 8; r++) w8[r] = wrowS[(rowbase + r) * 8 + e];

#pragma unroll
        for (int nc = 0; nc < 4; nc++) {
            // batch all 4 K-chunk B fragments, then chain 4 WMMAs
            v16h bf[4];
#pragma unroll
            for (int kc = 0; kc < 4; kc++) {
                int col = cg * 64 + nc * 16 + (lane & 15);
                const half_t* p = BtS + col * WT_STRIDE + kc * 32 + ((lane >> 4) << 3);
                v8h lo = *(const v8h*)p;
                v8h hi = *(const v8h*)(p + 16);
#pragma unroll
                for (int h = 0; h < 8; h++) { bf[kc][h] = lo[h]; bf[kc][h + 8] = hi[h]; }
            }
            v8f c;
#pragma unroll
            for (int q = 0; q < 8; q++) c[q] = 0.f;
#pragma unroll
            for (int kc = 0; kc < 4; kc++)
                c = __builtin_amdgcn_wmma_f32_16x16x32_f16(
                        false, afr[kc], false, bf[kc], (short)0, c, false, false);
            // relu then gate-weight per row, accumulate across experts
#pragma unroll
            for (int r = 0; r < 8; r++) acc[nc][r] += fmaxf(c[r], 0.f) * w8[r];
        }
        __syncthreads();          // tile j fully consumed before it is reused
    }

    // ---- stage weighted outputs [64][128] f32 (reuses weight buf 0) ----
#pragma unroll
    for (int nc = 0; nc < 4; nc++) {
        int colb = cg * 64 + nc * 16 + (lane & 15);
#pragma unroll
        for (int r = 0; r < 8; r++)
            outS[(rowbase + r) * 128 + colb] = acc[nc][r];
    }
    __syncthreads();

    // ---- segment-reduce the 64 sorted rows, flush via f32 global atomics ----
    int o  = tid & 127;
    int r0 = (tid >> 7) * 32;
    int cur = segsS[r0];
    float a = 0.f;
    for (int r = r0; r < r0 + 32; r++) {
        int s = segsS[r];
        if (s != cur) { atomicAdd(&y[(size_t)cur * ODIM + o], a); a = 0.f; cur = s; }
        a += outS[r * 128 + o];
    }
    atomicAdd(&y[(size_t)cur * ODIM + o], a);
}

// ---------------------------------------------------------------------------
extern "C" void kernel_launch(void* const* d_in, const int* in_sizes, int n_in,
                              void* d_out, int out_size, void* d_ws, size_t ws_size,
                              hipStream_t stream) {
    const float* feat   = (const float*)d_in[0];   // [N, D] f32
    const int*   seg    = (const int*)d_in[1];     // [N] i32 (sorted)
    const float* noise  = (const float*)d_in[2];   // [B, E] f32
    const float* wgate  = (const float*)d_in[3];   // [D, E] f32
    const float* wnoise = (const float*)d_in[4];   // [D, E] f32
    const float* wexp   = (const float*)d_in[5];   // [E, D, O] f32

    float* y     = (float*)d_out;                  // [B, O]
    float* gout  = y + (size_t)BGRAPHS * ODIM;     // [B, E]
    float* lossp = gout + (size_t)BGRAPHS * NEXP;  // scalar

    char* ws = (char*)d_ws;
    float*  fg    = (float*)(ws + FG_OFF);
    float*  gw    = (float*)(ws + GW_OFF);
    float*  imp   = (float*)(ws + IMP_OFF);
    float*  loadv = (float*)(ws + LOADV_OFF);
    half_t* wt    = (half_t*)(ws + WT_OFF);

    k_zero <<<512, 256, 0, stream>>>(fg, y, imp);
    k_prepw<<<512, 256, 0, stream>>>(wexp, wt);
    k_pool <<<NNODES / 128, 128, 0, stream>>>(feat, seg, fg);
    k_gate <<<4, 256, 0, stream>>>(fg, noise, wgate, wnoise, gout, gw, imp, loadv);
    k_loss <<<1, 32, 0, stream>>>(imp, loadv, lossp);
    k_expert<<<NNODES / 64, 256, SMEM_BYTES, stream>>>(feat, seg, gw, wt, y);
}
